// MultiHeadAttention_41128606827096
// MI455X (gfx1250) — compile-verified, matrix operand order checked by reading, not run on hardware
//
#include <hip/hip_runtime.h>
#include <hip/hip_bf16.h>

// ---------------------------------------------------------------------------
// MHA for MI455X (gfx1250): bf16 WMMA 16x16x32, f32 accumulate, software-
// pipelined fragment loads (double buffering with peeled epilogue).
// B=2, S=2048, D=1024, H=16, dk=64.
// ---------------------------------------------------------------------------

typedef __bf16 bf16_t;
typedef __attribute__((ext_vector_type(16))) __bf16 v16bf;
typedef __attribute__((ext_vector_type(8)))  __bf16 v8bf;
typedef __attribute__((ext_vector_type(4)))  __bf16 v4bf;
typedef __attribute__((ext_vector_type(8)))  float  v8f;
typedef __attribute__((ext_vector_type(4)))  float  v4f;

// ---- WMMA fragment helpers (layouts per CDNA5 ISA §7.12.2) ------

// A/B-fragment 16x32 bf16 from a row-major matrix (ld in elements).
// lane<16:  row=lane,     K = {0..7, 16..23}
// lane>=16: row=lane-16,  K = {8..15, 24..31}
__device__ __forceinline__ v16bf load_frag_bf16(const bf16_t* base, int ld, int lane) {
    int r  = lane & 15;
    int kf = (lane >> 4) << 3;              // 0 or 8
    const bf16_t* p = base + (size_t)r * ld + kf;
    v8bf lo = *(const v8bf*)(p);            // 16-byte load
    v8bf hi = *(const v8bf*)(p + 16);       // 16-byte load
    return __builtin_shufflevector(lo, hi, 0,1,2,3,4,5,6,7,8,9,10,11,12,13,14,15);
}

// Same fragment, source f32 row-major, converted to bf16 in-register.
__device__ __forceinline__ v16bf load_frag_f32(const float* base, int ld, int lane) {
    int r  = lane & 15;
    int kf = (lane >> 4) << 3;
    const float* p = base + (size_t)r * ld + kf;
    v8f lo = *(const v8f*)(p);              // 32-byte load
    v8f hi = *(const v8f*)(p + 16);
    v16bf a;
#pragma unroll
    for (int i = 0; i < 8; ++i) { a[i] = (bf16_t)lo[i]; a[i + 8] = (bf16_t)hi[i]; }
    return a;
}

__device__ __forceinline__ v8f wmma_bf16(v16bf a, v16bf b, v8f c) {
    return __builtin_amdgcn_wmma_f32_16x16x32_bf16(false, a, false, b, (short)0, c,
                                                   false, false);
}

// ---------------------------------------------------------------------------
// f32 -> bf16 conversion, 4 elements/thread
// ---------------------------------------------------------------------------
__global__ void cvt_f32_to_bf16_x4(const v4f* __restrict__ in,
                                   v4bf* __restrict__ out, int n4) {
    int i = blockIdx.x * blockDim.x + threadIdx.x;
    if (i < n4) {
        v4f x = in[i];
        v4bf y;
#pragma unroll
        for (int j = 0; j < 4; ++j) y[j] = (bf16_t)x[j];
        out[i] = y;
    }
}

// ---------------------------------------------------------------------------
// Projection GEMM:  C[m,n] = sum_k A[m,k] * W[n,k] + bias[n]
// One wave computes a 32x64 tile: 2 A-frags x 4 B-frags -> 8 WMMAs/k-step.
// Double-buffered fragment loads with a peeled, load-free epilogue.
// mode 0: bf16 row-major [M,N]            (Q, K projections)
// mode 1: bf16 per-batch transposed Vt:   vt[(b*N+n)*S + s]
// mode 2: f32 row-major [M,N]             (output projection)
// ---------------------------------------------------------------------------
__global__ void __launch_bounds__(32, 1)
gemm_bias_wmma(const bf16_t* __restrict__ A,
               const bf16_t* __restrict__ W,
               const float*  __restrict__ bias,
               bf16_t* __restrict__ outb,
               float*  __restrict__ outf,
               bf16_t* __restrict__ outvt,
               int M, int N, int K, int S, int mode) {
    const int lane   = threadIdx.x & 31;
    const int ntiles = N >> 6;
    const int mt = blockIdx.x / ntiles;
    const int nt = blockIdx.x - mt * ntiles;
    const int m0 = mt << 5;
    const int n0 = nt << 6;

    const bf16_t* A0 = A + (size_t)m0 * K;
    const bf16_t* A1 = A + (size_t)(m0 + 16) * K;
    const bf16_t* W0 = W + (size_t)n0 * K;
    const bf16_t* W1 = W0 + (size_t)16 * K;
    const bf16_t* W2 = W0 + (size_t)32 * K;
    const bf16_t* W3 = W0 + (size_t)48 * K;

    v8f cL0 = {}, cL1 = {}, cL2 = {}, cL3 = {};
    v8f cH0 = {}, cH1 = {}, cH2 = {}, cH3 = {};

    // preload k-step 0
    v16bf a0 = load_frag_bf16(A0, K, lane);
    v16bf a1 = load_frag_bf16(A1, K, lane);
    v16bf b0 = load_frag_bf16(W0, K, lane);
    v16bf b1 = load_frag_bf16(W1, K, lane);
    v16bf b2 = load_frag_bf16(W2, K, lane);
    v16bf b3 = load_frag_bf16(W3, K, lane);

    const int steps = K >> 5;
    for (int s = 0; s < steps - 1; ++s) {
        const int kn = (s + 1) << 5;
        v16bf na0 = load_frag_bf16(A0 + kn, K, lane);
        v16bf na1 = load_frag_bf16(A1 + kn, K, lane);
        v16bf nb0 = load_frag_bf16(W0 + kn, K, lane);
        v16bf nb1 = load_frag_bf16(W1 + kn, K, lane);
        v16bf nb2 = load_frag_bf16(W2 + kn, K, lane);
        v16bf nb3 = load_frag_bf16(W3 + kn, K, lane);

        cL0 = wmma_bf16(a0, b0, cL0);
        cL1 = wmma_bf16(a0, b1, cL1);
        cL2 = wmma_bf16(a0, b2, cL2);
        cL3 = wmma_bf16(a0, b3, cL3);
        cH0 = wmma_bf16(a1, b0, cH0);
        cH1 = wmma_bf16(a1, b1, cH1);
        cH2 = wmma_bf16(a1, b2, cH2);
        cH3 = wmma_bf16(a1, b3, cH3);

        a0 = na0; a1 = na1; b0 = nb0; b1 = nb1; b2 = nb2; b3 = nb3;
    }
    // epilogue: last k-step, no loads
    cL0 = wmma_bf16(a0, b0, cL0);
    cL1 = wmma_bf16(a0, b1, cL1);
    cL2 = wmma_bf16(a0, b2, cL2);
    cL3 = wmma_bf16(a0, b3, cL3);
    cH0 = wmma_bf16(a1, b0, cH0);
    cH1 = wmma_bf16(a1, b1, cH1);
    cH2 = wmma_bf16(a1, b2, cH2);
    cH3 = wmma_bf16(a1, b3, cH3);

    // C/D layout: VGPR i -> row = i + (lane>=16 ? 8 : 0), col = lane&15
    const int col   = lane & 15;
    const int rbase = (lane >> 4) << 3;
    v8f accL[4] = {cL0, cL1, cL2, cL3};
    v8f accH[4] = {cH0, cH1, cH2, cH3};
#pragma unroll
    for (int j = 0; j < 4; ++j) {
        const int   n  = n0 + j * 16 + col;
        const float bn = bias[n];
#pragma unroll
        for (int g = 0; g < 2; ++g) {
            const v8f* acc = g ? &accH[j] : &accL[j];
#pragma unroll
            for (int i = 0; i < 8; ++i) {
                float v = (*acc)[i] + bn;
                int m = m0 + g * 16 + rbase + i;
                if (mode == 0) {
                    outb[(size_t)m * N + n] = (bf16_t)v;
                } else if (mode == 2) {
                    outf[(size_t)m * N + n] = v;
                } else {  // Vt
                    int bb = m / S;
                    int s  = m - bb * S;
                    outvt[((size_t)bb * N + n) * S + s] = (bf16_t)v;
                }
            }
        }
    }
}

// ---------------------------------------------------------------------------
// Scores: attn_raw[b,h,q,k] = (Q . K) / 8.  One wave per (b, h, 32-row q-tile)
// sweeping all 128 key tiles; Q fragments held in registers the whole sweep.
// ---------------------------------------------------------------------------
__global__ void __launch_bounds__(32, 1)
scores_wmma(const bf16_t* __restrict__ Qb,
            const bf16_t* __restrict__ Kb,
            float* __restrict__ attn) {
    const int lane = threadIdx.x & 31;
    const int S = 2048, D = 1024;
    int qt = blockIdx.x & 63;            // S/32 = 64 q-tiles
    int h  = (blockIdx.x >> 6) & 15;
    int b  = blockIdx.x >> 10;

    const bf16_t* qbase = Qb + ((size_t)(b * S + qt * 32)) * D + h * 64;
    v16bf a00 = load_frag_bf16(qbase,                       D, lane);
    v16bf a01 = load_frag_bf16(qbase + 32,                  D, lane);
    v16bf a10 = load_frag_bf16(qbase + (size_t)16 * D,      D, lane);
    v16bf a11 = load_frag_bf16(qbase + (size_t)16 * D + 32, D, lane);

    float* arow = attn + ((size_t)((b * 16 + h) * S + qt * 32)) * S;
    const bf16_t* kbase = Kb + (size_t)b * S * D + h * 64;

    const int col   = lane & 15;
    const int rbase = (lane >> 4) << 3;

    v16bf b0 = load_frag_bf16(kbase,      D, lane);
    v16bf b1 = load_frag_bf16(kbase + 32, D, lane);
    for (int kt = 0; kt < 127; ++kt) {
        const bf16_t* nk = kbase + (size_t)((kt + 1) * 16) * D;
        v16bf nb0 = load_frag_bf16(nk,      D, lane);
        v16bf nb1 = load_frag_bf16(nk + 32, D, lane);

        v8f accL = {}, accH = {};
        accL = wmma_bf16(a00, b0, accL);
        accL = wmma_bf16(a01, b1, accL);
        accH = wmma_bf16(a10, b0, accH);
        accH = wmma_bf16(a11, b1, accH);

        float* dst = arow + kt * 16 + col;
#pragma unroll
        for (int i = 0; i < 8; ++i) {
            dst[(size_t)(rbase + i) * S]      = accL[i] * 0.125f;
            dst[(size_t)(16 + rbase + i) * S] = accH[i] * 0.125f;
        }
        b0 = nb0; b1 = nb1;
    }
    {   // epilogue kt = 127
        v8f accL = {}, accH = {};
        accL = wmma_bf16(a00, b0, accL);
        accL = wmma_bf16(a01, b1, accL);
        accH = wmma_bf16(a10, b0, accH);
        accH = wmma_bf16(a11, b1, accH);
        float* dst = arow + 127 * 16 + col;
#pragma unroll
        for (int i = 0; i < 8; ++i) {
            dst[(size_t)(rbase + i) * S]      = accL[i] * 0.125f;
            dst[(size_t)(16 + rbase + i) * S] = accH[i] * 0.125f;
        }
    }
}

// ---------------------------------------------------------------------------
// Masked softmax over one attn row of 2048, in place. 256 threads/row.
// ---------------------------------------------------------------------------
__global__ void softmax_mask(float* __restrict__ attn, const int* __restrict__ mask) {
    const int S = 2048;
    const int row = blockIdx.x;          // b*H*S + h*S + q
    const int q   = row & (S - 1);
    const int b   = (row / S) >> 4;
    float* a = attn + (size_t)row * S;
    const int* mrow = mask + ((size_t)(b * S + q)) * S;

    __shared__ float red[256];
    const int t = threadIdx.x;
    float v[8];
    float mx = -3.0e38f;
#pragma unroll
    for (int i = 0; i < 8; ++i) {
        int idx = t + i * 256;
        float s = a[idx];
        if (mrow[idx] == 0) s = -1e9f;
        v[i] = s;
        mx = fmaxf(mx, s);
    }
    red[t] = mx; __syncthreads();
    for (int off = 128; off > 0; off >>= 1) {
        if (t < off) red[t] = fmaxf(red[t], red[t + off]);
        __syncthreads();
    }
    mx = red[0]; __syncthreads();

    float sum = 0.f;
#pragma unroll
    for (int i = 0; i < 8; ++i) { v[i] = __expf(v[i] - mx); sum += v[i]; }
    red[t] = sum; __syncthreads();
    for (int off = 128; off > 0; off >>= 1) {
        if (t < off) red[t] += red[t + off];
        __syncthreads();
    }
    const float inv = 1.0f / red[0];
#pragma unroll
    for (int i = 0; i < 8; ++i) a[t + i * 256] = v[i] * inv;
}

// ---------------------------------------------------------------------------
// PV: x = attn @ V.  One wave per (b, h, 16-row q-tile); 16x64 output tile,
// double-buffered over the S (=K) dimension with peeled epilogue.
// Vt is [b, h*64+d, s].
// ---------------------------------------------------------------------------
__global__ void __launch_bounds__(32, 1)
pv_wmma(const float* __restrict__ attn,
        const bf16_t* __restrict__ Vt,
        bf16_t* __restrict__ xbf) {
    const int lane = threadIdx.x & 31;
    const int S = 2048, D = 1024;
    int qt = blockIdx.x & 127;
    int h  = (blockIdx.x >> 7) & 15;
    int b  = blockIdx.x >> 11;

    const float* arow = attn + ((size_t)((b * 16 + h) * S + qt * 16)) * S;
    const bf16_t* v0 = Vt + ((size_t)(b * D + h * 64)) * S;
    const bf16_t* v1 = v0 + (size_t)16 * S;
    const bf16_t* v2 = v0 + (size_t)32 * S;
    const bf16_t* v3 = v0 + (size_t)48 * S;

    v8f c0 = {}, c1 = {}, c2 = {}, c3 = {};

    v16bf a  = load_frag_f32(arow, S, lane);
    v16bf b0 = load_frag_bf16(v0, S, lane);
    v16bf b1 = load_frag_bf16(v1, S, lane);
    v16bf b2 = load_frag_bf16(v2, S, lane);
    v16bf b3 = load_frag_bf16(v3, S, lane);

    for (int k0 = 0; k0 < S - 32; k0 += 32) {
        const int kn = k0 + 32;
        v16bf na  = load_frag_f32(arow + kn, S, lane);
        v16bf nb0 = load_frag_bf16(v0 + kn, S, lane);
        v16bf nb1 = load_frag_bf16(v1 + kn, S, lane);
        v16bf nb2 = load_frag_bf16(v2 + kn, S, lane);
        v16bf nb3 = load_frag_bf16(v3 + kn, S, lane);

        c0 = wmma_bf16(a, b0, c0);
        c1 = wmma_bf16(a, b1, c1);
        c2 = wmma_bf16(a, b2, c2);
        c3 = wmma_bf16(a, b3, c3);

        a = na; b0 = nb0; b1 = nb1; b2 = nb2; b3 = nb3;
    }
    // epilogue: last k-step, no loads
    c0 = wmma_bf16(a, b0, c0);
    c1 = wmma_bf16(a, b1, c1);
    c2 = wmma_bf16(a, b2, c2);
    c3 = wmma_bf16(a, b3, c3);

    const int col   = lane & 15;
    const int rbase = (lane >> 4) << 3;
    v8f accs[4] = {c0, c1, c2, c3};
#pragma unroll
    for (int j = 0; j < 4; ++j)
#pragma unroll
        for (int i = 0; i < 8; ++i) {
            int m = b * S + qt * 16 + rbase + i;
            int n = h * 64 + j * 16 + col;
            xbf[(size_t)m * D + n] = (bf16_t)accs[j][i];
        }
}

// ---------------------------------------------------------------------------
// Launch
// ---------------------------------------------------------------------------
extern "C" void kernel_launch(void* const* d_in, const int* in_sizes, int n_in,
                              void* d_out, int out_size, void* d_ws, size_t ws_size,
                              hipStream_t stream) {
    const float* q  = (const float*)d_in[0];
    const float* k  = (const float*)d_in[1];
    const float* v  = (const float*)d_in[2];
    const int*   mk = (const int*)  d_in[3];
    const float* Wq = (const float*)d_in[4];
    const float* bq = (const float*)d_in[5];
    const float* Wk = (const float*)d_in[6];
    const float* bk = (const float*)d_in[7];
    const float* Wv = (const float*)d_in[8];
    const float* bv = (const float*)d_in[9];
    const float* Wo = (const float*)d_in[10];
    const float* bo = (const float*)d_in[11];

    const int B = 2, S = 2048, D = 1024, H = 16;
    const int M   = B * S;                 // 4096
    const int NEL = B * S * D;             // 4,194,304
    const int WEL = D * D;                 // 1,048,576

    float* out  = (float*)d_out;                        // [B,S,D]
    float* attn = out + (size_t)NEL;                    // [B,H,S,S]

    // workspace layout (bf16)
    char* ws = (char*)d_ws;
    bf16_t* q_bf  = (bf16_t*)(ws);
    bf16_t* k_bf  = q_bf  + (size_t)NEL;
    bf16_t* v_bf  = k_bf  + (size_t)NEL;
    bf16_t* wq_bf = v_bf  + (size_t)NEL;
    bf16_t* wk_bf = wq_bf + (size_t)WEL;
    bf16_t* wv_bf = wk_bf + (size_t)WEL;
    bf16_t* wo_bf = wv_bf + (size_t)WEL;
    bf16_t* Qb    = wo_bf + (size_t)WEL;   // [B,S,D]
    bf16_t* Kb    = Qb    + (size_t)NEL;   // [B,S,D]
    bf16_t* VtB   = Kb    + (size_t)NEL;   // [B, D, S]
    bf16_t* xbf   = VtB   + (size_t)NEL;   // [B,S,D]

    // 1) convert inputs + weights to bf16 (x4 vectorized)
    cvt_f32_to_bf16_x4<<<NEL / 4 / 256, 256, 0, stream>>>((const v4f*)q,  (v4bf*)q_bf,  NEL / 4);
    cvt_f32_to_bf16_x4<<<NEL / 4 / 256, 256, 0, stream>>>((const v4f*)k,  (v4bf*)k_bf,  NEL / 4);
    cvt_f32_to_bf16_x4<<<NEL / 4 / 256, 256, 0, stream>>>((const v4f*)v,  (v4bf*)v_bf,  NEL / 4);
    cvt_f32_to_bf16_x4<<<WEL / 4 / 256, 256, 0, stream>>>((const v4f*)Wq, (v4bf*)wq_bf, WEL / 4);
    cvt_f32_to_bf16_x4<<<WEL / 4 / 256, 256, 0, stream>>>((const v4f*)Wk, (v4bf*)wk_bf, WEL / 4);
    cvt_f32_to_bf16_x4<<<WEL / 4 / 256, 256, 0, stream>>>((const v4f*)Wv, (v4bf*)wv_bf, WEL / 4);
    cvt_f32_to_bf16_x4<<<WEL / 4 / 256, 256, 0, stream>>>((const v4f*)Wo, (v4bf*)wo_bf, WEL / 4);

    // 2) projections: grid = (M/32)*(D/64) = 128*16 = 2048 waves
    const int gemm_blocks = (M / 32) * (D / 64);
    gemm_bias_wmma<<<gemm_blocks, 32, 0, stream>>>(q_bf, wq_bf, bq, Qb, nullptr, nullptr,
                                                   M, D, D, S, 0);
    gemm_bias_wmma<<<gemm_blocks, 32, 0, stream>>>(k_bf, wk_bf, bk, Kb, nullptr, nullptr,
                                                   M, D, D, S, 0);
    gemm_bias_wmma<<<gemm_blocks, 32, 0, stream>>>(v_bf, wv_bf, bv, nullptr, nullptr, VtB,
                                                   M, D, D, S, 1);

    // 3) raw scores -> attn region of d_out
    scores_wmma<<<B * H * (S / 32), 32, 0, stream>>>(Qb, Kb, attn);

    // 4) masked softmax in place
    softmax_mask<<<B * H * S, 256, 0, stream>>>(attn, mk);

    // 5) attn @ V -> xbf (bf16)
    pv_wmma<<<B * H * (S / 16), 32, 0, stream>>>(attn, VtB, xbf);

    // 6) output projection -> d_out (f32)
    gemm_bias_wmma<<<gemm_blocks, 32, 0, stream>>>(xbf, wo_bf, bo, nullptr, out, nullptr,
                                                   M, D, D, S, 2);
}
